// ParticleAttentionBlock_78288663872455
// MI455X (gfx1250) — compile-verified, matrix operand order checked by reading, not run on hardware
//
#include <hip/hip_runtime.h>

typedef __attribute__((ext_vector_type(16))) __bf16 v16bf;
typedef __attribute__((ext_vector_type(8)))  float  v8f;

// ---------------- problem constants ----------------
constexpr int Bc = 16, Nc = 256, Dc = 256, Hc = 8, HDc = 32, PDc = 4;
constexpr int ROWS = Bc * Nc;           // 4096 token rows
constexpr int PAIRS = Bc * Nc * Nc;     // 1,048,576 pair rows

// ---------------- workspace layout (bytes) ----------------
constexpr size_t OFF_WT_QKV  = 0;                                  // 768x256 bf16
constexpr size_t OFF_WT_PROJ = OFF_WT_QKV  + 768ull*256*2;         // 256x256 bf16
constexpr size_t OFF_WT_MLP1 = OFF_WT_PROJ + 256ull*256*2;         // 1024x256 bf16
constexpr size_t OFF_WT_MLP2 = OFF_WT_MLP1 + 1024ull*256*2;        // 256x1024 bf16
constexpr size_t OFF_WT_PW1  = OFF_WT_MLP2 + 256ull*1024*2;        // 64x32 bf16 (K padded 4->32)
constexpr size_t OFF_WT_PW2  = OFF_WT_PW1  + 64ull*32*2;           // 64x64 bf16
constexpr size_t OFF_WT_PW3  = OFF_WT_PW2  + 64ull*64*2;           // 16x64 bf16 (N padded 8->16)
constexpr size_t OFF_XNORM   = OFF_WT_PW3  + 16ull*64*2;           // 4096x256 bf16
constexpr size_t OFF_QKV     = OFF_XNORM   + 4096ull*256*2;        // 4096x768 bf16
constexpr size_t OFF_BIAS    = OFF_QKV     + 4096ull*768*2;        // [B,H,N,N] f32
constexpr size_t OFF_XA      = OFF_BIAS    + 16ull*8*256*256*4;    // 4096x256 bf16
constexpr size_t OFF_Y       = OFF_XA      + 4096ull*256*2;        // 4096x256 f32
constexpr size_t OFF_YNORM   = OFF_Y       + 4096ull*256*4;        // 4096x256 bf16
constexpr size_t OFF_M1      = OFF_YNORM   + 4096ull*256*2;        // 4096x1024 bf16
constexpr size_t OFF_VT      = OFF_M1      + 4096ull*1024*2;       // [B,H,HD,N] bf16 (V^T)

// ---------------- helpers ----------------
static __device__ inline __bf16 f2bf(float f) {
    unsigned u = __builtin_bit_cast(unsigned, f);
    unsigned r = u + 0x7FFFu + ((u >> 16) & 1u);
    unsigned short h = (unsigned short)(r >> 16);
    return __builtin_bit_cast(__bf16, h);
}

// Branch-free exact-GELU: Abramowitz-Stegun 7.1.26 erf approx (|err| < 1.5e-7),
// straight-line v_fma chain + v_exp_f32 + v_rcp_f32 -- no EXEC-mask branching,
// no IEEE div expansion.
static __device__ inline float gelu_f(float x) {
    float z  = fabsf(x) * 0.70710678118654752440f;
    float t  = __builtin_amdgcn_rcpf(1.0f + 0.3275911f * z);   // v_rcp_f32
    float p  = t * (0.254829592f +
               t * (-0.284496736f +
               t * (1.421413741f +
               t * (-1.453152027f +
               t * 1.061405429f))));
    float er = 1.0f - p * __expf(-z * z);
    er = copysignf(er, x);
    return 0.5f * x * (1.0f + er);
}

static __device__ inline v8f wmma_bf16(v16bf a, v16bf b, v8f c) {
    return __builtin_amdgcn_wmma_f32_16x16x32_bf16(
        false, a, false, b, (short)0, c, false, false);
}

// Per-lane fragment loader for the ISA's 16-bit 16x32 A/B layout:
// lane half = lane>>4; lane holds K = {k0 + half*8 .. +7} and {k0+16+half*8 .. +7}
static __device__ inline v16bf load_frag(const __bf16* rowp, int k0, int half) {
    const __bf16* p0 = rowp + k0 + half * 8;
    const __bf16* p1 = p0 + 16;
    v16bf r;
#pragma unroll
    for (int j = 0; j < 8; ++j) { r[j] = p0[j]; r[8 + j] = p1[j]; }
    return r;
}

static __device__ inline float redmax16(float v) {
#pragma unroll
    for (int m = 1; m <= 8; m <<= 1) v = fmaxf(v, __shfl_xor(v, m, 32));
    return v;
}
static __device__ inline float redsum16(float v) {
#pragma unroll
    for (int m = 1; m <= 8; m <<= 1) v += __shfl_xor(v, m, 32);
    return v;
}

// ---------------- weight prep: f32 [K,N] -> bf16 [Npad,Kpad] transposed, zero pad ----------------
__global__ void prep_wt(const float* __restrict__ w, __bf16* __restrict__ wt,
                        int K, int N, int Kpad, int Npad) {
    int idx = blockIdx.x * blockDim.x + threadIdx.x;
    int total = Npad * Kpad;
    if (idx >= total) return;
    int n = idx / Kpad, k = idx % Kpad;
    float v = (k < K && n < N) ? w[(size_t)k * N + n] : 0.0f;
    wt[idx] = f2bf(v);
}

// ---------------- V transpose: qkv[B,N, 512+h*32+d] -> vt[B,H,HD,N] bf16 ----------------
__global__ void prep_vt(const __bf16* __restrict__ qkv, __bf16* __restrict__ vt) {
    int idx = blockIdx.x * blockDim.x + threadIdx.x;   // Bc*Hc*HDc*Nc = 1,048,576
    int m = idx & 255;
    int d = (idx >> 8) & 31;
    int h = (idx >> 13) & 7;
    int b = idx >> 16;
    vt[idx] = qkv[((size_t)b * Nc + m) * 768 + 512 + h * HDc + d];
}

// ---------------- LayerNorm: wave per row of 256, f32 in -> bf16 out ----------------
__global__ void ln_kernel(const float* __restrict__ x, const float* __restrict__ g,
                          const float* __restrict__ b, __bf16* __restrict__ out, int rows) {
    int wid  = (blockIdx.x * blockDim.x + threadIdx.x) >> 5;
    int lane = threadIdx.x & 31;
    if (wid >= rows) return;
    const float* xr = x + (size_t)wid * Dc;
    float v[8]; float s = 0.0f;
#pragma unroll
    for (int i = 0; i < 8; ++i) { v[i] = xr[i * 32 + lane]; s += v[i]; }
#pragma unroll
    for (int m = 1; m <= 16; m <<= 1) s += __shfl_xor(s, m, 32);
    float mu = s * (1.0f / Dc);
    float q = 0.0f;
#pragma unroll
    for (int i = 0; i < 8; ++i) { float d = v[i] - mu; q += d * d; }
#pragma unroll
    for (int m = 1; m <= 16; m <<= 1) q += __shfl_xor(q, m, 32);
    float inv = __builtin_amdgcn_rsqf(q * (1.0f / Dc) + 1e-5f);   // v_rsq_f32
#pragma unroll
    for (int i = 0; i < 8; ++i) {
        int c = i * 32 + lane;
        out[(size_t)wid * Dc + c] = f2bf((v[i] - mu) * inv * g[c] + b[c]);
    }
}

// ---------------- pairwise MLP: u[B,N,N,4] -> bias[B,H,N,N] ----------------
__global__ void pairwise_kernel(const float* __restrict__ u,
                                const __bf16* __restrict__ w1t, const float* __restrict__ b1,
                                const __bf16* __restrict__ w2t, const float* __restrict__ b2,
                                const __bf16* __restrict__ w3t, const float* __restrict__ b3,
                                float* __restrict__ biasws) {
    __shared__ __bf16 lds[8][16][64];
    int wave = threadIdx.x >> 5, lane = threadIdx.x & 31;
    int lrow = lane & 15, half = lane >> 4;
    int rowbase = blockIdx.x * 128 + wave * 16;

    // --- layer 1: [16,4(pad32)] @ [32,64] ---
    v16bf a;
#pragma unroll
    for (int j = 0; j < 16; ++j) a[j] = __builtin_bit_cast(__bf16, (unsigned short)0);
    if (half == 0) {
        const float* urow = u + (size_t)(rowbase + lrow) * PDc;
#pragma unroll
        for (int j = 0; j < PDc; ++j) a[j] = f2bf(urow[j]);
    }
    v8f c1[4];
#pragma unroll
    for (int t = 0; t < 4; ++t) {
        v8f z = {};
        v16bf bf = load_frag(w1t + (size_t)(t * 16 + lrow) * 32, 0, half);
        c1[t] = wmma_bf16(a, bf, z);
    }
#pragma unroll
    for (int t = 0; t < 4; ++t) {
        int col = t * 16 + lrow;
        float bb = b1[col];
#pragma unroll
        for (int r = 0; r < 8; ++r)
            lds[wave][r + 8 * half][col] = f2bf(gelu_f(c1[t][r] + bb));
    }
    __syncthreads();

    // --- layer 2: [16,64] @ [64,64] ---
    v8f c2[4] = {{}, {}, {}, {}};
#pragma unroll
    for (int kt = 0; kt < 2; ++kt) {
        v16bf a2 = load_frag(&lds[wave][lrow][0], kt * 32, half);
#pragma unroll
        for (int t = 0; t < 4; ++t) {
            v16bf bf = load_frag(w2t + (size_t)(t * 16 + lrow) * 64, kt * 32, half);
            c2[t] = wmma_bf16(a2, bf, c2[t]);
        }
    }
    __syncthreads();
#pragma unroll
    for (int t = 0; t < 4; ++t) {
        int col = t * 16 + lrow;
        float bb = b2[col];
#pragma unroll
        for (int r = 0; r < 8; ++r)
            lds[wave][r + 8 * half][col] = f2bf(gelu_f(c2[t][r] + bb));
    }
    __syncthreads();

    // --- layer 3: [16,64] @ [64,8(pad16)] ---
    v8f c3 = {};
#pragma unroll
    for (int kt = 0; kt < 2; ++kt) {
        v16bf a3 = load_frag(&lds[wave][lrow][0], kt * 32, half);
        v16bf bf = load_frag(w3t + (size_t)lrow * 64, kt * 32, half);
        c3 = wmma_bf16(a3, bf, c3);
    }
    if (lrow < Hc) {
        float bb = b3[lrow];
        // lane's 8 outputs are consecutive j at fixed (b,h,i): two float4 stores
        int p0 = rowbase + 8 * half;
        int b  = p0 >> 16;
        int i  = (p0 >> 8) & 255;
        int j0 = p0 & 255;
        float o[8];
#pragma unroll
        for (int r = 0; r < 8; ++r) o[r] = gelu_f(c3[r] + bb);
        float* dst = biasws + (((size_t)b * Hc + lrow) * Nc + i) * Nc + j0;
        *reinterpret_cast<float4*>(dst)     = make_float4(o[0], o[1], o[2], o[3]);
        *reinterpret_cast<float4*>(dst + 4) = make_float4(o[4], o[5], o[6], o[7]);
    }
}

// ---------------- generic GEMM: C = epilogue(A[M,K] @ WtT + bias) ----------------
// MODE 0: bf16 out; 1: gelu->bf16; 2: res + v -> f32; 3: res + gelu(v) -> f32
template <int MODE>
__global__ void gemm_kernel(const __bf16* __restrict__ A, int lda,
                            const __bf16* __restrict__ Wt,   // [N,K] bf16
                            const float* __restrict__ bias,
                            int M, int N, int K,
                            __bf16* __restrict__ out_bf,
                            float* __restrict__ out_f,
                            const float* __restrict__ res, int ldo) {
    int wave = threadIdx.x >> 5, lane = threadIdx.x & 31;
    int lrow = lane & 15, half = lane >> 4;
    int row0 = blockIdx.x * 128 + wave * 16;
    int col0 = blockIdx.y * 64;

    const __bf16* arow = A + (size_t)(row0 + lrow) * lda;
    v8f c[4] = {{}, {}, {}, {}};
    for (int k0 = 0; k0 < K; k0 += 32) {
        if (k0 + 32 < K) __builtin_prefetch(arow + k0 + 32, 0, 0);  // global_prefetch_b8
        v16bf a = load_frag(arow, k0, half);
#pragma unroll
        for (int t = 0; t < 4; ++t) {
            v16bf b = load_frag(Wt + (size_t)(col0 + t * 16 + lrow) * K, k0, half);
            c[t] = wmma_bf16(a, b, c[t]);
        }
    }
#pragma unroll
    for (int t = 0; t < 4; ++t) {
        int col = col0 + t * 16 + lrow;
        float bb = bias[col];
#pragma unroll
        for (int r = 0; r < 8; ++r) {
            int row = row0 + r + 8 * half;
            float v = c[t][r] + bb;
            if (MODE == 0) {
                out_bf[(size_t)row * ldo + col] = f2bf(v);
            } else if (MODE == 1) {
                out_bf[(size_t)row * ldo + col] = f2bf(gelu_f(v));
            } else if (MODE == 2) {
                out_f[(size_t)row * ldo + col] = res[(size_t)row * ldo + col] + v;
            } else {
                out_f[(size_t)row * ldo + col] = res[(size_t)row * ldo + col] + gelu_f(v);
            }
        }
    }
}

// ---------------- attention: scores(WMMA) + bias + softmax + P@V(WMMA) ----------------
// 4 waves/block (128 threads); wave owns 16 q rows x all 256 keys for one (b,h).
__global__ void attn_kernel(const __bf16* __restrict__ qkv,    // [4096, 768]
                            const __bf16* __restrict__ vt,     // [B,H,HD,N] bf16
                            const float* __restrict__ biasws,  // [B,H,N,N]
                            const unsigned char* __restrict__ mask, // [B,N]
                            __bf16* __restrict__ xa) {         // [4096, 256]
    __shared__ __bf16 plds[4][16][256];
    int wave = threadIdx.x >> 5, lane = threadIdx.x & 31;
    int lrow = lane & 15, half = lane >> 4;
    int strip = blockIdx.x & 3;
    int bh = blockIdx.x >> 2;
    int b = bh >> 3, h = bh & 7;
    int qrow0 = strip * 64 + wave * 16;
    const float scale = 0.17677669529663687f; // 1/sqrt(32)

    const __bf16* base = qkv + (size_t)b * Nc * 768 + (size_t)h * HDc;

    // q fragment: single K-tile (HD = 32)
    v16bf aq = load_frag(base + (size_t)(qrow0 + lrow) * 768, 0, half);

    // scores: 16 column tiles of 16
    v8f c[16];
#pragma unroll
    for (int t = 0; t < 16; ++t) {
        v8f z = {};
        v16bf kf = load_frag(base + 256 + (size_t)(t * 16 + lrow) * 768, 0, half);
        c[t] = wmma_bf16(aq, kf, z);
    }

    // scale + pairwise bias + mask
    const float* bbase = biasws + ((size_t)b * Hc + h) * Nc * Nc;
#pragma unroll
    for (int t = 0; t < 16; ++t) {
        int j = t * 16 + lrow;
        bool mk = mask[b * Nc + j] != 0;
#pragma unroll
        for (int r = 0; r < 8; ++r) {
            int i = qrow0 + r + 8 * half;
            float v = c[t][r] * scale + bbase[(size_t)i * Nc + j];
            c[t][r] = mk ? -1e30f : v;
        }
    }

    // row softmax (rows live across the 16 lanes of each half-wave)
#pragma unroll
    for (int r = 0; r < 8; ++r) {
        float m = -1e30f;
#pragma unroll
        for (int t = 0; t < 16; ++t) m = fmaxf(m, c[t][r]);
        m = redmax16(m);
        float s = 0.0f;
#pragma unroll
        for (int t = 0; t < 16; ++t) { float e = __expf(c[t][r] - m); c[t][r] = e; s += e; }
        s = redsum16(s);
        float inv = __builtin_amdgcn_rcpf(s);   // v_rcp_f32, no div expansion
#pragma unroll
        for (int t = 0; t < 16; ++t) c[t][r] *= inv;
    }

    // P -> LDS (bf16), re-layout C -> A fragments
#pragma unroll
    for (int t = 0; t < 16; ++t)
#pragma unroll
        for (int r = 0; r < 8; ++r)
            plds[wave][r + 8 * half][t * 16 + lrow] = f2bf(c[t][r]);
    __syncthreads();

    // xa = P @ V : K = 256 (8 tiles), N = 32 (2 tiles); V^T gives contiguous B frags
    v8f o[2] = {{}, {}};
    const __bf16* vtb = vt + (size_t)bh * HDc * Nc;
#pragma unroll
    for (int kt = 0; kt < 8; ++kt) {
        v16bf ap = load_frag(&plds[wave][lrow][0], kt * 32, half);
#pragma unroll
        for (int dt = 0; dt < 2; ++dt) {
            v16bf bf = load_frag(vtb + (size_t)(dt * 16 + lrow) * Nc, kt * 32, half);
            o[dt] = wmma_bf16(ap, bf, o[dt]);
        }
    }
#pragma unroll
    for (int dt = 0; dt < 2; ++dt)
#pragma unroll
        for (int r = 0; r < 8; ++r) {
            int row = qrow0 + r + 8 * half;
            int col = h * HDc + dt * 16 + lrow;
            xa[((size_t)b * Nc + row) * Dc + col] = f2bf(o[dt][r]);
        }
}

// ---------------- launch ----------------
extern "C" void kernel_launch(void* const* d_in, const int* in_sizes, int n_in,
                              void* d_out, int out_size, void* d_ws, size_t ws_size,
                              hipStream_t stream) {
    const float* x      = (const float*)d_in[0];
    const float* u      = (const float*)d_in[1];
    const unsigned char* mask = (const unsigned char*)d_in[2];
    const float* n1_g   = (const float*)d_in[3];
    const float* n1_b   = (const float*)d_in[4];
    const float* qkv_w  = (const float*)d_in[5];
    const float* qkv_b  = (const float*)d_in[6];
    const float* proj_w = (const float*)d_in[7];
    const float* proj_b = (const float*)d_in[8];
    const float* n2_g   = (const float*)d_in[9];
    const float* n2_b   = (const float*)d_in[10];
    const float* mlp_w1 = (const float*)d_in[11];
    const float* mlp_b1 = (const float*)d_in[12];
    const float* mlp_w2 = (const float*)d_in[13];
    const float* mlp_b2 = (const float*)d_in[14];
    const float* pw_w1  = (const float*)d_in[15];
    const float* pw_b1  = (const float*)d_in[16];
    const float* pw_w2  = (const float*)d_in[17];
    const float* pw_b2  = (const float*)d_in[18];
    const float* pw_w3  = (const float*)d_in[19];
    const float* pw_b3  = (const float*)d_in[20];

    char* ws = (char*)d_ws;
    __bf16* wt_qkv  = (__bf16*)(ws + OFF_WT_QKV);
    __bf16* wt_proj = (__bf16*)(ws + OFF_WT_PROJ);
    __bf16* wt_mlp1 = (__bf16*)(ws + OFF_WT_MLP1);
    __bf16* wt_mlp2 = (__bf16*)(ws + OFF_WT_MLP2);
    __bf16* wt_pw1  = (__bf16*)(ws + OFF_WT_PW1);
    __bf16* wt_pw2  = (__bf16*)(ws + OFF_WT_PW2);
    __bf16* wt_pw3  = (__bf16*)(ws + OFF_WT_PW3);
    __bf16* xnorm   = (__bf16*)(ws + OFF_XNORM);
    __bf16* qkvws   = (__bf16*)(ws + OFF_QKV);
    float*  biasws  = (float*)(ws + OFF_BIAS);
    __bf16* xaws    = (__bf16*)(ws + OFF_XA);
    float*  yws     = (float*)(ws + OFF_Y);
    __bf16* ynorm   = (__bf16*)(ws + OFF_YNORM);
    __bf16* m1ws    = (__bf16*)(ws + OFF_M1);
    __bf16* vtws    = (__bf16*)(ws + OFF_VT);

    auto cdiv = [](int a, int b) { return (a + b - 1) / b; };

    // weight prep (bf16, transposed, padded)
    prep_wt<<<cdiv(768 * 256, 256), 256, 0, stream>>>(qkv_w,  wt_qkv,  256, 768,  256, 768);
    prep_wt<<<cdiv(256 * 256, 256), 256, 0, stream>>>(proj_w, wt_proj, 256, 256,  256, 256);
    prep_wt<<<cdiv(1024 * 256, 256), 256, 0, stream>>>(mlp_w1, wt_mlp1, 256, 1024, 256, 1024);
    prep_wt<<<cdiv(256 * 1024, 256), 256, 0, stream>>>(mlp_w2, wt_mlp2, 1024, 256, 1024, 256);
    prep_wt<<<cdiv(64 * 32, 256), 256, 0, stream>>>(pw_w1, wt_pw1, PDc, 64, 32, 64);
    prep_wt<<<cdiv(64 * 64, 256), 256, 0, stream>>>(pw_w2, wt_pw2, 64, 64, 64, 64);
    prep_wt<<<cdiv(16 * 64, 256), 256, 0, stream>>>(pw_w3, wt_pw3, 64, Hc, 64, 16);

    // pairwise bias tensor
    pairwise_kernel<<<PAIRS / 128, 256, 0, stream>>>(u, wt_pw1, pw_b1, wt_pw2, pw_b2,
                                                     wt_pw3, pw_b3, biasws);

    // LN1 -> QKV
    ln_kernel<<<ROWS / 8, 256, 0, stream>>>(x, n1_g, n1_b, xnorm, ROWS);
    gemm_kernel<0><<<dim3(ROWS / 128, 768 / 64), 256, 0, stream>>>(
        xnorm, 256, wt_qkv, qkv_b, ROWS, 768, 256, qkvws, nullptr, nullptr, 768);

    // V transpose for contiguous P@V fragments
    prep_vt<<<(Bc * Hc * HDc * Nc) / 256, 256, 0, stream>>>(qkvws, vtws);

    // attention
    attn_kernel<<<Bc * Hc * 4, 128, 0, stream>>>(qkvws, vtws, biasws, mask, xaws);

    // projection + residual -> y (f32)
    gemm_kernel<2><<<dim3(ROWS / 128, 256 / 64), 256, 0, stream>>>(
        xaws, 256, wt_proj, proj_b, ROWS, 256, 256, nullptr, yws, x, 256);

    // LN2 -> MLP
    ln_kernel<<<ROWS / 8, 256, 0, stream>>>(yws, n2_g, n2_b, ynorm, ROWS);
    gemm_kernel<1><<<dim3(ROWS / 128, 1024 / 64), 256, 0, stream>>>(
        ynorm, 256, wt_mlp1, mlp_b1, ROWS, 1024, 256, m1ws, nullptr, nullptr, 1024);
    gemm_kernel<3><<<dim3(ROWS / 128, 256 / 64), 256, 0, stream>>>(
        m1ws, 1024, wt_mlp2, mlp_b2, ROWS, 256, 1024, nullptr, (float*)d_out, yws, 256);
}
